// AAGRU_53687091200307
// MI455X (gfx1250) — compile-verified, matrix operand order checked by reading, not run on hardware
//
#include <hip/hip_runtime.h>
#include <hip/hip_bf16.h>
#include <math.h>

#define NB   64
#define TT   1024
#define FEAT 256
#define HID  256
#define G3   768
#define H2   512

typedef __attribute__((ext_vector_type(16))) __bf16       v16bf;
typedef __attribute__((ext_vector_type(2)))  __bf16       bf2;
typedef __attribute__((ext_vector_type(8)))  float        v8f;
typedef __attribute__((ext_vector_type(2)))  float        vf2;
typedef __attribute__((ext_vector_type(8)))  unsigned int v8u;
typedef __attribute__((ext_vector_type(4)))  int          v4i;

// Async global->LDS (CDNA5): guard so the file compiles on any toolchain.
#if defined(__AMDGCN__) && defined(__gfx1250__) && \
    __has_builtin(__builtin_amdgcn_global_load_async_to_lds_b128) && \
    __has_builtin(__builtin_amdgcn_s_wait_asynccnt)
#define USE_ASYNC_LDS 1
typedef __attribute__((address_space(1))) v4i* gv4i_p;   // global int4*
typedef __attribute__((address_space(3))) v4i* lv4i_p;   // LDS    int4*
#else
#define USE_ASYNC_LDS 0
#endif

// fp32x2 -> packed bf16x2 (hardware cvt, RNE)
__device__ __forceinline__ unsigned int packbf(float lo, float hi) {
    vf2 f; f.x = lo; f.y = hi;
    bf2 p = __builtin_convertvector(f, bf2);
    return __builtin_bit_cast(unsigned int, p);
}

__device__ __forceinline__ void stage16(float* dst, const float* __restrict__ src) {
#if USE_ASYNC_LDS
    __builtin_amdgcn_global_load_async_to_lds_b128((gv4i_p)src, (lv4i_p)dst, 0, 0);
#else
    *(float4*)dst = *(const float4*)src;
#endif
}

__device__ __forceinline__ void wait_async() {
#if USE_ASYNC_LDS
    __builtin_amdgcn_s_wait_asynccnt(0);
#endif
}

// ---------------------------------------------------------------------------
// Pre-pack a (256 x 768) fp32 weight slab into bf16 WMMA B-fragments.
// dst[((ct*8 + kt)*32 + lane)*8 + v] holds bf16{W[k0][col], W[k0+1][col]}
// with col = colOff + ct*16 + lane%16, K pattern per ISA 16-bit layout.
// ---------------------------------------------------------------------------
__global__ void prepack_bf16(const float* __restrict__ w, int colOff,
                             int nColTiles, unsigned int* __restrict__ dst) {
    int idx = blockIdx.x * blockDim.x + threadIdx.x;
    int total = nColTiles * 8 * 256;
    if (idx >= total) return;
    int ct   = idx >> 11;
    int rem  = idx & 2047;
    int kt   = rem >> 8;
    int q    = rem & 255;
    int lane = q >> 3;
    int v    = q & 7;
    int col   = colOff + ct * 16 + (lane & 15);
    int kbase = kt * 32 + ((lane < 16) ? 0 : 8);
    int k0    = (v < 4) ? (kbase + 2 * v) : (kbase + 16 + 2 * (v - 4));
    dst[idx] = packbf(w[(size_t)k0 * G3 + col], w[(size_t)(k0 + 1) * G3 + col]);
}

// ---------------------------------------------------------------------------
// Gate precompute: G[t][n][g] = x[n,t]@w_i + a[n,t]@w_a + b  (fp32, time-major
// so the scan reads one contiguous 48 KB slice per step).
// One 16x16 output tile per wave, K=256 as 8 bf16 WMMAs of K=32.
// ---------------------------------------------------------------------------
__global__ void __launch_bounds__(256)
gates_gemm(const float* __restrict__ x, const float* __restrict__ a,
           const float* __restrict__ wa, const float* __restrict__ b,
           const unsigned int* __restrict__ wi_pack, float* __restrict__ Gg) {
    int wid  = blockIdx.x * (blockDim.x >> 5) + (threadIdx.x >> 5);
    int lane = threadIdx.x & 31;
    int rt = wid / 48;               // row tile over N*T = 65536 rows
    int ct = wid % 48;               // col tile over 768 gate cols
    int nt0  = rt * 16;
    int m16  = lane & 15;
    int koff = (lane < 16) ? 0 : 8;
    int krow = nt0 + m16;            // flattened row this lane supplies for A

    v8f acc = {};
    #pragma unroll
    for (int kt = 0; kt < 8; ++kt) {
        const float* xp = x + (size_t)krow * FEAT + kt * 32 + koff;
        float4 a0 = *(const float4*)(xp + 0);
        float4 a1 = *(const float4*)(xp + 4);
        float4 a2 = *(const float4*)(xp + 16);
        float4 a3 = *(const float4*)(xp + 20);
        v8u au;
        au[0] = packbf(a0.x, a0.y); au[1] = packbf(a0.z, a0.w);
        au[2] = packbf(a1.x, a1.y); au[3] = packbf(a1.z, a1.w);
        au[4] = packbf(a2.x, a2.y); au[5] = packbf(a2.z, a2.w);
        au[6] = packbf(a3.x, a3.y); au[7] = packbf(a3.z, a3.w);
        const unsigned int* wp = wi_pack + (((size_t)ct * 8 + kt) * 32 + lane) * 8;
        uint4 b0 = *(const uint4*)(wp);
        uint4 b1 = *(const uint4*)(wp + 4);
        v8u bu;
        bu[0] = b0.x; bu[1] = b0.y; bu[2] = b0.z; bu[3] = b0.w;
        bu[4] = b1.x; bu[5] = b1.y; bu[6] = b1.z; bu[7] = b1.w;
        acc = __builtin_amdgcn_wmma_f32_16x16x32_bf16(
            false, __builtin_bit_cast(v16bf, au),
            false, __builtin_bit_cast(v16bf, bu),
            (short)0, acc, false, false);
    }

    int col = ct * 16 + m16;
    float wac[8];
    #pragma unroll
    for (int j = 0; j < 8; ++j) wac[j] = wa[(size_t)j * G3 + col];
    float bcol = b[col];
    int mofs = (lane < 16) ? 0 : 8;
    #pragma unroll
    for (int v = 0; v < 8; ++v) {
        int row = nt0 + v + mofs;           // flattened n*T + t
        int n = row >> 10, t = row & (TT - 1);
        const float* ap = a + (size_t)row * 8;
        float4 av0 = *(const float4*)ap;
        float4 av1 = *(const float4*)(ap + 4);
        float dot = av0.x * wac[0] + av0.y * wac[1] + av0.z * wac[2] + av0.w * wac[3]
                  + av1.x * wac[4] + av1.y * wac[5] + av1.z * wac[6] + av1.w * wac[7];
        Gg[((size_t)t * NB + n) * G3 + col] = acc[v] + dot + bcol;   // time-major
    }
}

// ---------------------------------------------------------------------------
// Sequential GRU scan. 4 workgroups x 16 batch rows; h lives in LDS.
// Gate slice for step t+1 is async-prefetched (ASYNCcnt) into a double
// buffer while step t computes. Weights stream from L2 in WMMA-fragment
// order with explicit prefetch.
// ---------------------------------------------------------------------------
__global__ void __launch_bounds__(512)
aagru_scan(const float* __restrict__ Gg,
           const unsigned int* __restrict__ whz_pack,
           const unsigned int* __restrict__ wha_pack,
           const unsigned char* __restrict__ reset,
           const float* __restrict__ ih, float* __restrict__ out) {
    __shared__ float hS[16][HID];
    __shared__ float zS[16][HID];
    __shared__ float rhS[16][HID];
    __shared__ __align__(16) unsigned int hbf[16][HID / 2];
    __shared__ __align__(16) unsigned int rhbf[16][HID / 2];
    extern __shared__ __align__(16) float gbuf[];   // 2 x 16*768 floats (96 KB)

    const int tid  = threadIdx.x;
    const int lane = tid & 31;
    const int wave = tid >> 5;
    const int n0   = blockIdx.x * 16;
    const int m16  = lane & 15;
    const int mofs = (lane < 16) ? 0 : 8;
    const size_t NTH = (size_t)NB * TT * HID;
    const int SLICE = 16 * G3;                      // floats per step slice

    // h := initial_h ; stage gates for t=0
    for (int i = tid; i < 16 * HID; i += 512)
        hS[i >> 8][i & 255] = ih[i & 255];
    {
        const float* src = Gg + (size_t)n0 * G3;    // t = 0 slice
        for (int i = tid * 4; i < SLICE; i += 512 * 4)
            stage16(gbuf + i, src + i);
    }
    __syncthreads();

    for (int t = 0; t < TT; ++t) {
        float* gcur = gbuf + (t & 1) * SLICE;

        // pass0: apply reset, pack h to bf16
        for (int i = tid; i < 16 * (HID / 2); i += 512) {
            int m = i >> 7, c2 = i & 127;
            float h0 = hS[m][2 * c2], h1 = hS[m][2 * c2 + 1];
            if (reset[(size_t)(n0 + m) * TT + t]) {
                h0 = ih[2 * c2]; h1 = ih[2 * c2 + 1];
                hS[m][2 * c2] = h0; hS[m][2 * c2 + 1] = h1;
            }
            hbf[m][c2] = packbf(h0, h1);
        }
        wait_async();              // gate slice for t is fully in LDS
        __syncthreads();

        // kick off async prefetch of gate slice for t+1
        if (t + 1 < TT) {
            const float* src = Gg + ((size_t)(t + 1) * NB + n0) * G3;
            float* dst = gbuf + ((t + 1) & 1) * SLICE;
            for (int i = tid * 4; i < SLICE; i += 512 * 4)
                stage16(dst + i, src + i);
        }

        // matmul1: zr = G_z + h @ Whz, sigmoid; 32 col tiles over 512 cols
        #pragma unroll
        for (int half = 0; half < 2; ++half) {
            int ctl = wave + half * 16;
            int col = ctl * 16 + m16;
            v8f acc;
            #pragma unroll
            for (int v = 0; v < 8; ++v)
                acc[v] = gcur[(v + mofs) * G3 + col];
            #pragma unroll
            for (int kt = 0; kt < 8; ++kt) {
                int abase = kt * 16 + ((lane < 16) ? 0 : 4);
                uint4 ha0 = *(const uint4*)&hbf[m16][abase];
                uint4 ha1 = *(const uint4*)&hbf[m16][abase + 8];
                v8u au;
                au[0] = ha0.x; au[1] = ha0.y; au[2] = ha0.z; au[3] = ha0.w;
                au[4] = ha1.x; au[5] = ha1.y; au[6] = ha1.z; au[7] = ha1.w;
                const unsigned int* wp =
                    whz_pack + (((size_t)ctl * 8 + kt) * 32 + lane) * 8;
                if (kt < 7) __builtin_prefetch(wp + 256, 0, 1);
                uint4 b0 = *(const uint4*)wp;
                uint4 b1 = *(const uint4*)(wp + 4);
                v8u bu;
                bu[0] = b0.x; bu[1] = b0.y; bu[2] = b0.z; bu[3] = b0.w;
                bu[4] = b1.x; bu[5] = b1.y; bu[6] = b1.z; bu[7] = b1.w;
                acc = __builtin_amdgcn_wmma_f32_16x16x32_bf16(
                    false, __builtin_bit_cast(v16bf, au),
                    false, __builtin_bit_cast(v16bf, bu),
                    (short)0, acc, false, false);
            }
            #pragma unroll
            for (int v = 0; v < 8; ++v) {
                int m = v + mofs;
                float s = 1.0f / (1.0f + __expf(-acc[v]));
                if (ctl < 16) {
                    zS[m][col] = s;
                } else {
                    int c = col - HID;
                    rhS[m][c] = s * hS[m][c];
                }
            }
        }
        __syncthreads();

        // pack r*h to bf16
        for (int i = tid; i < 16 * (HID / 2); i += 512) {
            int m = i >> 7, c2 = i & 127;
            rhbf[m][c2] = packbf(rhS[m][2 * c2], rhS[m][2 * c2 + 1]);
        }
        __syncthreads();

        // matmul2: cand = tanh(G_a + (r*h) @ Wha); blend; store
        {
            int ctl = wave;
            int col = ctl * 16 + m16;
            v8f acc;
            #pragma unroll
            for (int v = 0; v < 8; ++v)
                acc[v] = gcur[(v + mofs) * G3 + H2 + col];
            #pragma unroll
            for (int kt = 0; kt < 8; ++kt) {
                int abase = kt * 16 + ((lane < 16) ? 0 : 4);
                uint4 ha0 = *(const uint4*)&rhbf[m16][abase];
                uint4 ha1 = *(const uint4*)&rhbf[m16][abase + 8];
                v8u au;
                au[0] = ha0.x; au[1] = ha0.y; au[2] = ha0.z; au[3] = ha0.w;
                au[4] = ha1.x; au[5] = ha1.y; au[6] = ha1.z; au[7] = ha1.w;
                const unsigned int* wp =
                    wha_pack + (((size_t)ctl * 8 + kt) * 32 + lane) * 8;
                if (kt < 7) __builtin_prefetch(wp + 256, 0, 1);
                uint4 b0 = *(const uint4*)wp;
                uint4 b1 = *(const uint4*)(wp + 4);
                v8u bu;
                bu[0] = b0.x; bu[1] = b0.y; bu[2] = b0.z; bu[3] = b0.w;
                bu[4] = b1.x; bu[5] = b1.y; bu[6] = b1.z; bu[7] = b1.w;
                acc = __builtin_amdgcn_wmma_f32_16x16x32_bf16(
                    false, __builtin_bit_cast(v16bf, au),
                    false, __builtin_bit_cast(v16bf, bu),
                    (short)0, acc, false, false);
            }
            #pragma unroll
            for (int v = 0; v < 8; ++v) {
                int m = v + mofs;
                float at = tanhf(acc[v]);
                float z  = zS[m][col];
                float hn = (1.0f - z) * hS[m][col] + z * at;
                hS[m][col] = hn;
                size_t o = ((size_t)(n0 + m) * TT + t) * HID + col;
                out[o]       = hn;   // outputs
                out[NTH + o] = hn;   // states (identical)
            }
        }
        __syncthreads();
    }
}

__global__ void write_ih(const float* __restrict__ ih, float* __restrict__ out) {
    int i = threadIdx.x;
    if (i < HID) out[(size_t)2 * NB * TT * HID + i] = ih[i];
}

extern "C" void kernel_launch(void* const* d_in, const int* in_sizes, int n_in,
                              void* d_out, int out_size, void* d_ws, size_t ws_size,
                              hipStream_t stream) {
    (void)in_sizes; (void)n_in; (void)out_size; (void)ws_size;
    const float*         x     = (const float*)d_in[0];
    const float*         a     = (const float*)d_in[1];
    const unsigned char* reset = (const unsigned char*)d_in[2];  // jnp bool (1B)
    const float*         w_i   = (const float*)d_in[3];
    const float*         w_h   = (const float*)d_in[4];
    const float*         w_a   = (const float*)d_in[5];
    const float*         b     = (const float*)d_in[6];
    const float*         ih    = (const float*)d_in[7];
    float*               out   = (float*)d_out;

    // workspace: fp32 gates (time-major, 192 MB) then packed bf16 weights
    float* Gg = (float*)d_ws;
    size_t Ge = (size_t)NB * TT * G3;
    unsigned int* wi_pack  = (unsigned int*)(Gg + Ge);   // 48 col tiles
    unsigned int* whz_pack = wi_pack + 48 * 2048;        // 32 col tiles
    unsigned int* wha_pack = whz_pack + 32 * 2048;       // 16 col tiles

    prepack_bf16<<<(48 * 2048) / 256, 256, 0, stream>>>(w_i, 0,   48, wi_pack);
    prepack_bf16<<<(32 * 2048) / 256, 256, 0, stream>>>(w_h, 0,   32, whz_pack);
    prepack_bf16<<<(16 * 2048) / 256, 256, 0, stream>>>(w_h, 512, 16, wha_pack);

    // 65536/16 row tiles * 48 col tiles = 196608 wave tiles, 8 waves/block
    gates_gemm<<<196608 / 8, 256, 0, stream>>>(x, a, w_a, b, wi_pack, Gg);

    size_t dynLds = (size_t)2 * 16 * G3 * sizeof(float);  // 96 KB double buffer
    aagru_scan<<<NB / 16, 512, dynLds, stream>>>(Gg, whz_pack, wha_pack,
                                                 reset, ih, out);
    write_ih<<<1, 256, 0, stream>>>(ih, out);
}
